// TemporalDecoderLSTM_61065845014979
// MI455X (gfx1250) — compile-verified, matrix operand order checked by reading, not run on hardware
//
#include <hip/hip_runtime.h>
#include <hip/hip_bf16.h>

#define OBS 20
#define PRED 30
#define HD 128
#define GD 512            // 4*HD
#define KCAT 160          // 16 (emb) + 128 (h) + 16 pad -> 5 K-slices of 32
#define NSLICE 5
#define NTILE 32          // GD / 16
#define WAVES 4
#define AGW 16            // agents per wave
#define AGB (WAVES*AGW)   // agents per block = 64

typedef __attribute__((ext_vector_type(16))) _Float16 v16h;
typedef __attribute__((ext_vector_type(8)))  _Float16 v8h;
typedef __attribute__((ext_vector_type(8)))  float    v8f;

// wave-local LDS fence: all comms (ring/acat) are intra-wave, so a dscnt
// wait is enough -- no block-wide barrier needed in the hot loop.
__device__ __forceinline__ void wave_lds_fence() {
  asm volatile("s_wait_dscnt 0x0" ::: "memory");
}

// fast activations: v_exp_f32 is exp2 on AMD; rcp instead of full divide
__device__ __forceinline__ float fsig(float x) {
  return __builtin_amdgcn_rcpf(1.f + __builtin_amdgcn_exp2f(-1.44269504089f * x));
}
__device__ __forceinline__ float ftanh(float x) {
  // 2/(1+exp2(-2*log2e*x)) - 1
  return fmaf(2.f, __builtin_amdgcn_rcpf(1.f + __builtin_amdgcn_exp2f(-2.88539008178f * x)), -1.f);
}

// ---------------------------------------------------------------------------
// Prep: pack concat(W_ih, W_hh) into WMMA B-fragment order, f16.
//   n = 16*t + (lane&15),  K = 32*s + (j>>3)*16 + (lane>>4)*8 + (j&7)
// ---------------------------------------------------------------------------
__global__ void pack_weights_kernel(const float* __restrict__ Wih,
                                    const float* __restrict__ Whh,
                                    _Float16* __restrict__ wpack) {
  int idx = blockIdx.x * 256 + threadIdx.x;           // total NTILE*NSLICE*32*16
  if (idx >= NTILE * NSLICE * 32 * 16) return;
  int j    = idx & 15;
  int lane = (idx >> 4) & 31;
  int s    = (idx >> 9) % NSLICE;
  int t    = idx / (NSLICE * 512);
  int n    = 16 * t + (lane & 15);
  int K    = 32 * s + ((j >> 3) * 16) + ((lane >> 4) * 8) + (j & 7);
  float v = 0.f;
  if (K < 16)        v = Wih[n * 16 + K];
  else if (K < 144)  v = Whh[n * HD + (K - 16)];
  wpack[idx] = (_Float16)v;
}

// ---------------------------------------------------------------------------
// Main: persistent recurrent decoder. 1 block = 64 agents, 4 waves x 32 lanes.
// ---------------------------------------------------------------------------
__launch_bounds__(128, 1)
__global__ void lstm_decoder_kernel(
    const float* __restrict__ traj_rel, const float* __restrict__ h0,
    const float* __restrict__ c0,
    const float* __restrict__ ln1w_g, const float* __restrict__ ln1b_g,
    const float* __restrict__ embW_g, const float* __restrict__ embB_g,
    const float* __restrict__ bih,  const float* __restrict__ bhh,
    const float* __restrict__ ln2w_g, const float* __restrict__ ln2b_g,
    const float* __restrict__ outW_g, const float* __restrict__ outB_g,
    const _Float16* __restrict__ wpack,
    float* __restrict__ out, int N) {

  extern __shared__ _Float16 smem_w[];               // [NTILE][NSLICE][32][16] = 160 KB
  const v16h* __restrict__ wfrag = (const v16h*)smem_w;

  __shared__ float s_bgate[GD];
  // LN1 folded into embedding:  di = tanh(rs*(x@A.T - mu*sumA) + C)
  __shared__ float s_embA[16][40];
  __shared__ float s_embSum[16];
  __shared__ float s_embC[16];
  // LN2 folded into output head: out = rs2*(h@A.T - mu2*sumA) + C
  __shared__ float s_outA[2][HD];
  __shared__ float s_outSum[2];
  __shared__ float s_outC[2];
  __shared__ float s_ring[WAVES][OBS][AGW][2];       // rolling rel-pos window
  __shared__ __align__(16) _Float16 s_acat[WAVES][AGW][KCAT]; // [di | h | pad] f16

  const int tid  = threadIdx.x;
  const int wave = tid >> 5;
  const int lane = tid & 31;
  const int g    = lane >> 4;                        // half-wave group
  const int lm   = lane & 15;
  const int agbase = blockIdx.x * AGB + wave * AGW;

  // ---- stage packed weights global -> LDS (once per block) ----
  {
    const uint4* src = (const uint4*)wpack;
    uint4* dst = (uint4*)smem_w;
    const int nvec = NTILE * NSLICE * 32 * 16 * (int)sizeof(_Float16) / 16;
    for (int i = tid; i < nvec; i += 128) dst[i] = src[i];
  }
  for (int i = tid; i < GD; i += 128) s_bgate[i] = bih[i] + bhh[i];
  if (tid < 16) {                                    // fold LN1 affine into embW
    float sA = 0.f, sC = 0.f;
    for (int j = 0; j < 40; ++j) {
      float w = embW_g[tid * 40 + j];
      float a = ln1w_g[j] * w;
      s_embA[tid][j] = a;
      sA += a;
      sC += ln1b_g[j] * w;
    }
    s_embSum[tid] = sA;
    s_embC[tid]   = sC + embB_g[tid];
  }
  if (tid < 2) {                                     // fold LN2 affine into outW
    float sA = 0.f, sC = 0.f;
    for (int k = 0; k < HD; ++k) {
      float w = outW_g[tid * HD + k];
      float a = ln2w_g[k] * w;
      s_outA[tid][k] = a;
      sA += a;
      sC += ln2b_g[k] * w;
    }
    s_outSum[tid] = sA;
    s_outC[tid]   = sC + outB_g[tid];
  }

  // ---- init trajectory ring ----
  for (int i = lane; i < OBS * AGW * 2; i += 32) {
    int k = i / (AGW * 2);
    int r = i % (AGW * 2);
    s_ring[wave][k][r >> 1][r & 1] =
        traj_rel[(size_t)k * N * 2 + (size_t)(agbase + (r >> 1)) * 2 + (r & 1)];
  }
  // ---- init h (as f16 in A staging), zero di + K-pad regions ----
  for (int i = lane; i < AGW * HD; i += 32) {
    int a = i >> 7, k = i & 127;
    s_acat[wave][a][16 + k] = (_Float16)h0[(size_t)(agbase + a) * HD + k];
  }
  for (int i = lane; i < AGW * 16; i += 32) {
    int a = i >> 4, k = i & 15;
    s_acat[wave][a][k]       = (_Float16)0.f;
    s_acat[wave][a][144 + k] = (_Float16)0.f;
  }
  // ---- init c in WMMA C-fragment layout (regs): agent = r + 8*g, feat = 16u+lm
  float cfr[8][8];
#pragma unroll
  for (int u = 0; u < 8; ++u)
#pragma unroll
    for (int r = 0; r < 8; ++r)
      cfr[u][r] = c0[(size_t)(agbase + r + 8 * g) * HD + 16 * u + lm];

  __syncthreads();   // weights + folded params visible to all waves

#pragma unroll 1
  for (int st = 0; st < PRED; ++st) {
    // ---- LN1 stats: lane covers j = 20*g..20*g+19 of agent lm, xor-combine
    float sm = 0.f, sq = 0.f;
#pragma unroll
    for (int j2 = 0; j2 < 20; ++j2) {
      int j = 20 * g + j2;
      float x = s_ring[wave][(st + (j >> 1)) % OBS][lm][j & 1];
      sm += x; sq += x * x;
    }
    sm += __shfl_xor(sm, 16, 32);
    sq += __shfl_xor(sq, 16, 32);
    float mu1 = sm * (1.f / 40.f);
    float rs1 = __builtin_amdgcn_rsqf(fmaf(-mu1, mu1, sq * (1.f / 40.f)) + 1e-5f);

    // ---- embedding: di[agent=lm][e = 8g..8g+7] (LN folded into s_embA)
    {
      float acc[8];
#pragma unroll
      for (int e = 0; e < 8; ++e) acc[e] = 0.f;
      for (int j = 0; j < 40; ++j) {
        float x = s_ring[wave][(st + (j >> 1)) % OBS][lm][j & 1];
#pragma unroll
        for (int e = 0; e < 8; ++e) acc[e] = fmaf(x, s_embA[8 * g + e][j], acc[e]);
      }
#pragma unroll
      for (int e = 0; e < 8; ++e) {
        float v = fmaf(rs1, fmaf(-mu1, s_embSum[8 * g + e], acc[e]), s_embC[8 * g + e]);
        s_acat[wave][lm][8 * g + e] = (_Float16)ftanh(v);
      }
    }
    wave_lds_fence();   // di visible to all lanes of this wave

    // ---- build A fragments: 16x32 f16 per slice; lane lm = agent row
    v16h afr[NSLICE];
#pragma unroll
    for (int s = 0; s < NSLICE; ++s) {
      v8h lo = *(const v8h*)&s_acat[wave][lm][32 * s + 8 * g];
      v8h hi = *(const v8h*)&s_acat[wave][lm][32 * s + 16 + 8 * g];
#pragma unroll
      for (int e = 0; e < 8; ++e) { afr[s][e] = lo[e]; afr[s][8 + e] = hi[e]; }
    }

    // ---- gates GEMM (16x512, K=160) + fused LSTM cell, 16-col blocks ----
#pragma unroll
    for (int u = 0; u < 8; ++u) {
      v8f acc[4];
      float bb[4];
#pragma unroll
      for (int q = 0; q < 4; ++q) {                  // q: i,f,g,o
        int t = 8 * q + u;                           // gate n-tile
        bb[q] = s_bgate[16 * t + lm];
        v8f c = {};                                  // zero C (bias added later)
#pragma unroll
        for (int s = 0; s < NSLICE; ++s) {
          v16h b = wfrag[(t * NSLICE + s) * 32 + lane];
          c = __builtin_amdgcn_wmma_f32_16x16x32_f16(
                  false, afr[s], false, b, (short)0, c, false, false);
        }
        acc[q] = c;
      }
#pragma unroll
      for (int r = 0; r < 8; ++r) {
        float iv = fsig(acc[0][r] + bb[0]);
        float fv = fsig(acc[1][r] + bb[1]);
        float gv = ftanh(acc[2][r] + bb[2]);
        float ov = fsig(acc[3][r] + bb[3]);
        float cn = fmaf(fv, cfr[u][r], iv * gv);
        cfr[u][r] = cn;
        float hn = ov * ftanh(cn);
        s_acat[wave][r + 8 * g][16 + 16 * u + lm] = (_Float16)hn;  // new h
      }
    }
    wave_lds_fence();   // new h visible to all lanes of this wave

    // ---- LN2 + output head (folded): lane -> (agent lm, output dim g) ----
    float sm2 = 0.f, sq2 = 0.f;
#pragma unroll
    for (int kk = 0; kk < 64; kk += 8) {
      v8h hv = *(const v8h*)&s_acat[wave][lm][16 + 64 * g + kk];
#pragma unroll
      for (int i = 0; i < 8; ++i) {
        float h = (float)hv[i];
        sm2 += h; sq2 += h * h;
      }
    }
    sm2 += __shfl_xor(sm2, 16, 32);
    sq2 += __shfl_xor(sq2, 16, 32);
    float mu2 = sm2 * (1.f / 128.f);
    float rs2 = __builtin_amdgcn_rsqf(fmaf(-mu2, mu2, sq2 * (1.f / 128.f)) + 1e-5f);

    float dacc = 0.f;
#pragma unroll
    for (int kk = 0; kk < HD; kk += 8) {
      v8h hv = *(const v8h*)&s_acat[wave][lm][16 + kk];
#pragma unroll
      for (int i = 0; i < 8; ++i)
        dacc = fmaf((float)hv[i], s_outA[g][kk + i], dacc);
    }
    float res = fmaf(rs2, fmaf(-mu2, s_outSum[g], dacc), s_outC[g]);

    out[(size_t)(agbase + lm) * (PRED * 2) + st * 2 + g] = res;    // [N,30,2]
    s_ring[wave][st % OBS][lm][g] = res;                           // roll window
    wave_lds_fence();   // ring update visible before next step's stats
  }
}

// ---------------------------------------------------------------------------
extern "C" void kernel_launch(void* const* d_in, const int* in_sizes, int n_in,
                              void* d_out, int out_size, void* d_ws, size_t ws_size,
                              hipStream_t stream) {
  // setup_inputs order: traj_abs(unused), traj_rel, h0, c0, ln1_w, ln1_b,
  // emb_W, emb_b, W_ih, W_hh, b_ih, b_hh, ln2_w, ln2_b, out_W, out_b
  const float* traj_rel = (const float*)d_in[1];
  const float* h0   = (const float*)d_in[2];
  const float* c0   = (const float*)d_in[3];
  const float* ln1w = (const float*)d_in[4];
  const float* ln1b = (const float*)d_in[5];
  const float* embW = (const float*)d_in[6];
  const float* embB = (const float*)d_in[7];
  const float* Wih  = (const float*)d_in[8];
  const float* Whh  = (const float*)d_in[9];
  const float* bih  = (const float*)d_in[10];
  const float* bhh  = (const float*)d_in[11];
  const float* ln2w = (const float*)d_in[12];
  const float* ln2b = (const float*)d_in[13];
  const float* outW = (const float*)d_in[14];
  const float* outB = (const float*)d_in[15];
  const int N = in_sizes[2] / HD;                   // h0 is [N,128]

  _Float16* wpack = (_Float16*)d_ws;                // 160 KB packed weights
  const int total = NTILE * NSLICE * 32 * 16;
  pack_weights_kernel<<<(total + 255) / 256, 256, 0, stream>>>(Wih, Whh, wpack);

  const size_t dynLDS = (size_t)total * sizeof(_Float16);  // 163840 bytes
  lstm_decoder_kernel<<<N / AGB, WAVES * 32, dynLDS, stream>>>(
      traj_rel, h0, c0, ln1w, ln1b, embW, embB, bih, bhh,
      ln2w, ln2b, outW, outB, wpack, (float*)d_out, N);
}
